// Base_40793599378196
// MI455X (gfx1250) — matrix operand order repaired and checked by reading, not compiled, not run on hardware
//
#include <hip/hip_runtime.h>
#include <hip/hip_bf16.h>

typedef __attribute__((ext_vector_type(16))) _Float16 v16h;
typedef __attribute__((ext_vector_type(8)))  float    v8f;

#define NNODES 65536   // B*NPG
#define HDIM   128
#define NEDGES 524288
#define BGRAPH 512
#define NPGC   128
#define NHF    ((long)NNODES * HDIM)   // 8388608
#define AROWF  36                      // f32 per A-tile LDS row (144B, 16B aligned)
#define WROWH  40                      // halves per W-tile LDS row (80B, 16B aligned)

// ---------------------------------------------------------------- utilities
__global__ void zero_f32(float* __restrict__ p, long n) {
  long i = (long)blockIdx.x * blockDim.x + threadIdx.x;
  long st = (long)gridDim.x * blockDim.x;
  for (; i < n; i += st) p[i] = 0.f;
}

// ---------------------------------------- LDS-staged batched WMMA GEMM
// C[z,m,n] = act( sum_k A[z,m,k]*W[z,k,n] + bias[z,n] )
// 256 threads = 8 waves; wave computes a 16x64 C strip (4 accumulators).
// A tile copied raw (f32) memory->LDS with GLOBAL_LOAD_ASYNC_TO_LDS_B128
// (ASYNCcnt), converted f32->f16 at fragment build. W tile staged manually
// (needs transpose + cvt) as f16. All 4 B fragments are loaded before the
// 4 back-to-back WMMAs so the XDL pipe streams.
__global__ __launch_bounds__(256)
void wmma_gemm_lds(const float* __restrict__ A, long aRow, long aZ,
                   const float* __restrict__ W, int ldw, long wZ,
                   const float* __restrict__ bias, int bZ,
                   float* __restrict__ C, long cRow, long cZ,
                   int K, int Nc, int doRelu) {
  const int tid  = threadIdx.x;
  const int lane = tid & 31;
  const int wv   = tid >> 5;
  const int wpn  = Nc >> 6;            // waves per N-row: 1 (Nc=64) or 2 (Nc=128)
  const int waveN = wv % wpn;
  const int waveM = wv / wpn;
  const int MBLK  = 128 / wpn;         // 128 or 64 rows per block
  const long rowBase = (long)blockIdx.x * MBLK;
  const int z = blockIdx.y;
  const int halfId = lane >> 4;        // 0: lanes 0-15, 1: lanes 16-31
  const int l16    = lane & 15;

  __shared__ __align__(16) float     Ashf[128 * AROWF];  // 18 KB (f32)
  __shared__ __align__(16) _Float16  Wsh [128 * WROWH];  // 10 KB (f16, transposed)

  const float* Ab = A + (long)z * aZ;
  const float* Wb = W + (long)z * wZ;

  v8f acc[4] = {{}, {}, {}, {}};
  const int aChunks = MBLK * 8;        // 16B chunks for A k-tile (MBLK x 32 f32)
  const int nc4     = Nc >> 2;
  const int wElems4 = 32 * nc4;        // float4 loads for W k-tile (32 x Nc)

  const float*    apf = &Ashf[(waveM * 16 + l16) * AROWF + halfId * 8];
  const _Float16* wpf = &Wsh[(waveN * 64 + l16) * WROWH + halfId * 8];

  for (int k0 = 0; k0 < K; k0 += 32) {
    __syncthreads();
    // ---- stage A (MBLK x 32 f32) via async global->LDS DMA, raw bytes
    for (int idx = tid; idx < aChunks; idx += 256) {
      const int row = idx >> 3;
      const int c   = (idx & 7) << 2;                 // f32 column
      const unsigned ldsOff =
          (unsigned)(size_t)(const void*)&Ashf[row * AROWF + c];
      const unsigned long long ga =
          (unsigned long long)(size_t)(Ab + (rowBase + row) * aRow + k0 + c);
      asm volatile("global_load_async_to_lds_b128 %0, %1, off"
                   :: "v"(ldsOff), "v"(ga) : "memory");
    }
    // ---- stage W (32 x Nc) f32 -> f16, transposed: Wsh[n][k]
    for (int idx = tid; idx < wElems4; idx += 256) {
      const int k  = idx / nc4;
      const int n0 = (idx % nc4) << 2;
      const float4 v = *(const float4*)(Wb + (long)(k0 + k) * ldw + n0);
      Wsh[(n0 + 0) * WROWH + k] = (_Float16)v.x;
      Wsh[(n0 + 1) * WROWH + k] = (_Float16)v.y;
      Wsh[(n0 + 2) * WROWH + k] = (_Float16)v.z;
      Wsh[(n0 + 3) * WROWH + k] = (_Float16)v.w;
    }
    asm volatile("s_wait_asynccnt 0x0" ::: "memory");  // async copies landed
    __syncthreads();

    // ---- A fragment: row waveM*16+l16, K = {0..7,16..23}+8*halfId (cvt here)
    union V16 { v16h v; uint4 q[2]; };
    const float4* aq = (const float4*)apf;
    const float4 al0 = aq[0], al1 = aq[1];         // k = kbA .. kbA+7
    const float4 ah0 = aq[4], ah1 = aq[5];         // k = kbA+16 .. kbA+23 (+64B)
    v16h a;
    a[0]  = (_Float16)al0.x; a[1]  = (_Float16)al0.y;
    a[2]  = (_Float16)al0.z; a[3]  = (_Float16)al0.w;
    a[4]  = (_Float16)al1.x; a[5]  = (_Float16)al1.y;
    a[6]  = (_Float16)al1.z; a[7]  = (_Float16)al1.w;
    a[8]  = (_Float16)ah0.x; a[9]  = (_Float16)ah0.y;
    a[10] = (_Float16)ah0.z; a[11] = (_Float16)ah0.w;
    a[12] = (_Float16)ah1.x; a[13] = (_Float16)ah1.y;
    a[14] = (_Float16)ah1.z; a[15] = (_Float16)ah1.w;

    // ---- load ALL 4 B fragments, then 4 back-to-back WMMAs
    V16 b[4];
#pragma unroll
    for (int t = 0; t < 4; ++t) {
      const _Float16* wp = wpf + t * 16 * WROWH;
      b[t].q[0] = *(const uint4*)(wp);             // k = kbB .. kbB+7
      b[t].q[1] = *(const uint4*)(wp + 16);        // k = kbB+16 .. kbB+23
    }
#pragma unroll
    for (int t = 0; t < 4; ++t) {
      acc[t] = __builtin_amdgcn_wmma_f32_16x16x32_f16(
          false, a, false, b[t].v, (short)0, acc[t], false, false);
    }
  }

  // ---- epilogue: bias + optional ReLU, store f32
#pragma unroll
  for (int t = 0; t < 4; ++t) {
    const int ccol = waveN * 64 + t * 16 + l16;
    const float bv = bias ? bias[(long)z * bZ + ccol] : 0.f;
#pragma unroll
    for (int r = 0; r < 8; ++r) {
      const long crow = rowBase + waveM * 16 + r + halfId * 8;
      float v = acc[t][r] + bv;
      if (doRelu) v = fmaxf(v, 0.f);
      C[(long)z * cZ + crow * cRow + ccol] = v;
    }
  }
}

// ------------------------------------------------------------- edge scatter
// agg[dst[e], :] += xn[src[e], :]   (xn & agg ~33.5MB each -> L2-resident)
__global__ void scatter_add_kernel(const float* __restrict__ xn,
                                   const int* __restrict__ src,
                                   const int* __restrict__ dst,
                                   float* __restrict__ agg, long E) {
  long t  = (long)blockIdx.x * blockDim.x + threadIdx.x;
  long st = (long)gridDim.x * blockDim.x;
  const long total = E * (HDIM / 4);
  for (; t < total; t += st) {
    const long e = t >> 5;                 // H/4 = 32 chunks of float4
    const int  c = (int)(t & 31) * 4;
    __builtin_prefetch(src + e + 8192, 0, 1);   // global_prefetch_b8
    __builtin_prefetch(dst + e + 8192, 0, 1);
    const int  s = src[e], d = dst[e];
    const float4 v = *(const float4*)(xn + (long)s * HDIM + c);
    float* p = agg + (long)d * HDIM + c;
    atomicAdd(p + 0, v.x); atomicAdd(p + 1, v.y);
    atomicAdd(p + 2, v.z); atomicAdd(p + 3, v.w);
  }
}

// -------------------------------------------- pre-activation + BN statistics
__global__ __launch_bounds__(128)
void pre_stats_kernel(float* __restrict__ xr, const float* __restrict__ agg,
                      const float* __restrict__ cb,
                      float* __restrict__ sums, float* __restrict__ sqs,
                      int rowsPerBlock) {
  const int f = threadIdx.x;                     // 0..127
  const long r0 = (long)blockIdx.x * rowsPerBlock;
  const float cbf = cb[f];
  float s = 0.f, q = 0.f;
  for (int r = 0; r < rowsPerBlock; ++r) {
    const long idx = (r0 + r) * HDIM + f;
    const float v = xr[idx] + agg[idx] + cbf;
    xr[idx] = v;
    s += v; q += v * v;
  }
  atomicAdd(&sums[f], s);
  atomicAdd(&sqs[f], q);
}

__global__ void bn_finalize_kernel(const float* __restrict__ sums,
                                   const float* __restrict__ sqs,
                                   const float* __restrict__ g,
                                   const float* __restrict__ b,
                                   float* __restrict__ scale,
                                   float* __restrict__ shift) {
  const int f = threadIdx.x;
  const float inv = 1.f / (float)NNODES;
  const float mean = sums[f] * inv;
  const float var  = sqs[f] * inv - mean * mean;
  const float sc   = g[f] * rsqrtf(var + 1e-5f);
  scale[f] = sc;
  shift[f] = b[f] - mean * sc;
}

__global__ void bn_apply_kernel(const float* __restrict__ pre,
                                const float* __restrict__ scale,
                                const float* __restrict__ shift,
                                float* __restrict__ h) {
  long t  = (long)blockIdx.x * blockDim.x + threadIdx.x;
  long st = (long)gridDim.x * blockDim.x;
  for (; t < NHF; t += st) {
    const int f = (int)(t & (HDIM - 1));
    h[t] = fmaxf(pre[t] * scale[f] + shift[f], 0.f);
  }
}

// ------------------------------------------------------ mean pool + ReLU
__global__ __launch_bounds__(128)
void pool_relu_kernel(const float* __restrict__ h, float* __restrict__ xg) {
  const int b = blockIdx.x, f = threadIdx.x;
  const float* p = h + (long)b * NPGC * HDIM + f;
  float s = 0.f;
  for (int n = 0; n < NPGC; ++n) s += p[(long)n * HDIM];
  xg[b * HDIM + f] = fmaxf(s * (1.f / NPGC), 0.f);
}

// --------------------------------------------------------- small tail GEMMs
__global__ void ghead_final_kernel(const float* __restrict__ A,
                                   const float* __restrict__ W,
                                   const float* __restrict__ bias,
                                   float* __restrict__ out) {
  const int t = blockIdx.x * blockDim.x + threadIdx.x;   // 512*2
  if (t >= BGRAPH * 2) return;
  const int b = t >> 1, o = t & 1;
  const float* a = A + (long)b * 64;
  float s = bias[o];
  for (int k = 0; k < 64; ++k) s += a[k] * W[k * 2 + o];
  out[t] = s;
}

__global__ void nhead_final_kernel(const float* __restrict__ nz2,
                                   const float* __restrict__ w3,
                                   const float* __restrict__ b3,
                                   float* __restrict__ nout) {
  const int t = blockIdx.x * blockDim.x + threadIdx.x;   // 65536
  if (t >= NNODES) return;
  const int n = t & (NPGC - 1);
  const float* a = nz2 + (long)t * 64;
  const float* w = w3 + (long)n * 64;
  float s = b3[n];
  for (int k = 0; k < 64; ++k) s += a[k] * w[k];
  nout[t] = s;
}

__global__ void write_out_kernel(const float* __restrict__ gfin,
                                 const float* __restrict__ nout,
                                 float* __restrict__ out) {
  const int t = blockIdx.x * blockDim.x + threadIdx.x;   // 512*130
  if (t >= BGRAPH * 130) return;
  const int b = t / 130, j = t % 130;
  out[t] = (j < 2) ? gfin[b * 2 + j] : nout[b * NPGC + (j - 2)];
}

// ---------------------------------------------------------------- host side
static inline dim3 gemm_grid(int M, int Nc, int z) {
  const int wpn = Nc >> 6;          // 1 or 2
  const int MBLK = 128 / wpn;       // 128 or 64
  return dim3(M / MBLK, z);
}

static void run_conv_layer(const float* hin, int K,
                           const float* wr, const float* wn, const float* cb,
                           const float* bg, const float* bb,
                           float* hout, float* xr, float* xn, float* agg,
                           float* stats, const int* src, const int* dst,
                           hipStream_t stream) {
  zero_f32<<<2048, 256, 0, stream>>>(agg, NHF);
  zero_f32<<<1, 256, 0, stream>>>(stats, 256);
  // xr = hin @ wr ; xn = hin @ wn   (no bias, no relu)
  wmma_gemm_lds<<<gemm_grid(NNODES, HDIM, 1), 256, 0, stream>>>(
      hin, K, 0, wr, HDIM, 0, nullptr, 0, xr, HDIM, 0, K, HDIM, 0);
  wmma_gemm_lds<<<gemm_grid(NNODES, HDIM, 1), 256, 0, stream>>>(
      hin, K, 0, wn, HDIM, 0, nullptr, 0, xn, HDIM, 0, K, HDIM, 0);
  scatter_add_kernel<<<4096, 256, 0, stream>>>(xn, src, dst, agg, NEDGES);
  pre_stats_kernel<<<NNODES / 64, 128, 0, stream>>>(xr, agg, cb,
                                                    stats, stats + 128, 64);
  bn_finalize_kernel<<<1, 128, 0, stream>>>(stats, stats + 128, bg, bb,
                                            stats + 256, stats + 384);
  bn_apply_kernel<<<2048, 256, 0, stream>>>(xr, stats + 256, stats + 384, hout);
}

extern "C" void kernel_launch(void* const* d_in, const int* in_sizes, int n_in,
                              void* d_out, int out_size, void* d_ws, size_t ws_size,
                              hipStream_t stream) {
  const float* x        = (const float*)d_in[0];
  const int*   edge     = (const int*)d_in[1];
  const int*   src      = edge;
  const int*   dst      = edge + NEDGES;
  const float* conv0_wr = (const float*)d_in[3];
  const float* conv0_wn = (const float*)d_in[4];
  const float* conv0_b  = (const float*)d_in[5];
  const float* bn0_g    = (const float*)d_in[6];
  const float* bn0_b    = (const float*)d_in[7];
  const float* convs_wr = (const float*)d_in[8];
  const float* convs_wn = (const float*)d_in[9];
  const float* convs_b  = (const float*)d_in[10];
  const float* bns_g    = (const float*)d_in[11];
  const float* bns_b    = (const float*)d_in[12];
  const float* gs_w1 = (const float*)d_in[13]; const float* gs_b1 = (const float*)d_in[14];
  const float* gs_w2 = (const float*)d_in[15]; const float* gs_b2 = (const float*)d_in[16];
  const float* gh_w1 = (const float*)d_in[17]; const float* gh_b1 = (const float*)d_in[18];
  const float* gh_w2 = (const float*)d_in[19]; const float* gh_b2 = (const float*)d_in[20];
  const float* gh_w3 = (const float*)d_in[21]; const float* gh_b3 = (const float*)d_in[22];
  const float* nh_w1 = (const float*)d_in[23]; const float* nh_b1 = (const float*)d_in[24];
  const float* nh_w2 = (const float*)d_in[25]; const float* nh_b2 = (const float*)d_in[26];
  const float* nh_w3 = (const float*)d_in[27]; const float* nh_b3 = (const float*)d_in[28];
  float* out = (float*)d_out;

  // workspace layout (floats); ~135 MB total
  float* ws    = (float*)d_ws;
  float* hbuf  = ws;                         // N*H
  float* xr    = hbuf + NHF;                 // N*H (also BN pre-activation)
  float* xn    = xr + NHF;                   // N*H (reused as nz1)
  float* agg   = xn + NHF;                   // N*H (reused as nz2: N*64)
  float* stats = agg + NHF;                  // 512: sums,sqs,scale,shift
  float* xg    = stats + 512;                // B*H
  float* g1    = xg + (long)BGRAPH * HDIM;   // B*H
  float* g2    = g1 + (long)BGRAPH * HDIM;   // B*H
  float* g4    = g2 + (long)BGRAPH * HDIM;   // B*64
  float* gfin  = g4 + (long)BGRAPH * 64;     // B*2
  float* nout  = gfin + (long)BGRAPH * 2;    // B*NPG

  // ---- 3 GCN + BN + ReLU layers (node-side GEMM -> edge scatter)
  run_conv_layer(x, 32, conv0_wr, conv0_wn, conv0_b, bn0_g, bn0_b,
                 hbuf, xr, xn, agg, stats, src, dst, stream);
  for (int i = 0; i < 2; ++i) {
    run_conv_layer(hbuf, HDIM,
                   convs_wr + (long)i * HDIM * HDIM,
                   convs_wn + (long)i * HDIM * HDIM,
                   convs_b + (long)i * HDIM,
                   bns_g + (long)i * HDIM, bns_b + (long)i * HDIM,
                   hbuf, xr, xn, agg, stats, src, dst, stream);
  }

  // ---- graph branch: mean-pool (+ReLU), then MLP chain (M=512)
  pool_relu_kernel<<<BGRAPH, 128, 0, stream>>>(hbuf, xg);
  wmma_gemm_lds<<<gemm_grid(BGRAPH, HDIM, 1), 256, 0, stream>>>(
      xg, HDIM, 0, gs_w1, HDIM, 0, gs_b1, 0, g1, HDIM, 0, HDIM, HDIM, 0);
  wmma_gemm_lds<<<gemm_grid(BGRAPH, HDIM, 1), 256, 0, stream>>>(
      g1, HDIM, 0, gs_w2, HDIM, 0, gs_b2, 0, g2, HDIM, 0, HDIM, HDIM, 1);
  wmma_gemm_lds<<<gemm_grid(BGRAPH, HDIM, 1), 256, 0, stream>>>(
      g2, HDIM, 0, gh_w1, HDIM, 0, gh_b1, 0, g1, HDIM, 0, HDIM, HDIM, 1);
  wmma_gemm_lds<<<gemm_grid(BGRAPH, 64, 1), 256, 0, stream>>>(
      g1, HDIM, 0, gh_w2, 64, 0, gh_b2, 0, g4, 64, 0, HDIM, 64, 1);
  ghead_final_kernel<<<4, 256, 0, stream>>>(g4, gh_w3, gh_b3, gfin);

  // ---- node branch: 128 batched GEMMs (blockIdx.y = node-in-graph index)
  wmma_gemm_lds<<<gemm_grid(BGRAPH, HDIM, NPGC), 256, 0, stream>>>(
      hbuf, (long)NPGC * HDIM, HDIM,               // A rows strided by NPG*H
      nh_w1, HDIM, (long)HDIM * HDIM,
      nh_b1, HDIM,
      xn, (long)NPGC * HDIM, HDIM,                 // nz1 (reuses xn)
      HDIM, HDIM, 1);
  wmma_gemm_lds<<<gemm_grid(BGRAPH, 64, NPGC), 256, 0, stream>>>(
      xn, (long)NPGC * HDIM, HDIM,
      nh_w2, 64, (long)HDIM * 64,
      nh_b2, 64,
      agg, (long)NPGC * 64, 64,                    // nz2 (reuses agg)
      HDIM, 64, 1);
  nhead_final_kernel<<<NNODES / 256, 256, 0, stream>>>(agg, nh_w3, nh_b3, nout);

  // ---- assemble (512, 130) output
  write_out_kernel<<<(BGRAPH * 130 + 255) / 256, 256, 0, stream>>>(gfin, nout, out);
}
